// EquivWeightQueryModelDeprecated_4114578670366
// MI455X (gfx1250) — matrix operand order repaired and checked by reading, not compiled, not run on hardware
//
#include <hip/hip_runtime.h>
#include <hip/hip_bf16.h>

#define N_PTS    4096
#define SH_DIM   16
#define NB       10
#define EMB      86
#define N_QUERY  64
#define K_MSG    12
#define K_RG     16
#define K_FLD    24

#define MAX_RADIUS_F   0.09f
#define QUERY_RADIUS_F 0.08f
#define FIELD_CUTOFF_F 0.12f
#define PI_F 3.14159265358979323846f
#define INV_SQRT_EMB 0.10783277320343841f   /* 1/sqrt(86) */

typedef __attribute__((ext_vector_type(16))) _Float16 v16h;
typedef __attribute__((ext_vector_type(2)))  _Float16 v2h;
typedef __attribute__((ext_vector_type(8)))  float    v8f;

__device__ __forceinline__ float wave_sum(float v) {
  #pragma unroll
  for (int o = 16; o > 0; o >>= 1) v += __shfl_xor(v, o, 32);
  return v;
}

// ---------------------------------------------------------------------------
// WMMA 16x16x32 f16 fragment layouts (wave32, per CDNA5 ISA):
//   A (16x32): lane L holds row M=L&15 (hs=L>>4); element j maps to
//              K = (j<8 ? 0:16) + hs*8 + (j&7)   [pairs packed 2/VGPR]
//   inverse:   for (row r, K=kk): lane = r + ((kk>>3)&1)*16,
//              elem j = (kk&7) | ((kk&16)>>1)
//   B (32x16): lane L -> column L&15; element j -> K same as A (hs=L>>4)
//   C/D f32:   elem r -> row r + (L>=16 ? 8 : 0), col L&15
// A-tiles staged in LDS pre-swizzled -> one 32B LDS read per lane; filled with
// packed 2xf16 stores (adjacent even/odd K share lane-row and adjacent j).
// B weights pre-swizzled in HBM -> one 32B global read per lane + prefetch.
// ---------------------------------------------------------------------------

// Write one packed A-tile pair. p in [0,256): row r=p>>4, K pair kk=2*(p&15).
__device__ __forceinline__ void a_fill_pair(_Float16 (*As)[16],
                                            const float (*xn_s)[EMB],
                                            const float (*sh_s)[SH_DIM],
                                            int ch, int p) {
  int r = p >> 4, kk = (p & 15) << 1;
  int c = (ch << 5) + kk;
  int f = c >> 4, s = c & 15;          // c even -> s even, s+1 <= 15
  float xv = xn_s[r][f];
  v2h pk;
  pk.x = (_Float16)(xv * sh_s[r][s]);
  pk.y = (_Float16)(xv * sh_s[r][s + 1]);
  int lanei = r + (((kk >> 3) & 1) << 4);
  int j0 = (kk & 7) | ((kk & 16) >> 1);  // even
  *(v2h*)(&As[lanei][j0]) = pk;
}

// Pre-swizzle f32 weights [CD x Dout] into WMMA B-fragment order:
// dst[(((tile*nch)+ch)*32 + lane)*16 + j] = W[c*Dout + col]  (zero padded)
__global__ void swizzle_b_kernel(const float* __restrict__ W, _Float16* __restrict__ dst,
                                 int CD, int Dout, int nTile, int nch) {
  int i = blockIdx.x * blockDim.x + threadIdx.x;
  int total = nTile * nch * 32 * 16;
  if (i >= total) return;
  int j    = i & 15;
  int lane = (i >> 4) & 31;
  int blk  = i >> 9;            // tile*nch + ch
  int ch   = blk % nch;
  int tile = blk / nch;
  int c   = ch * 32 + ((lane >> 4) * 8) + (j & 7) + ((j >> 3) << 4);
  int col = tile * 16 + (lane & 15);
  float v = (c < CD && col < Dout) ? W[(size_t)c * Dout + col] : 0.f;
  dst[i] = (_Float16)v;
}

// ---------------------------------------------------------------------------
// Brute-force top-k radius graph (k <= 24), one thread per destination point.
// ---------------------------------------------------------------------------
#define KNN_MAX 24
__global__ void knn_kernel(const float* __restrict__ pd, const float* __restrict__ ps,
                           int nd, int ns, int k, float rmax, int exclude_self,
                           int* __restrict__ idx, int* __restrict__ msk,
                           float* __restrict__ dst, float* __restrict__ unt) {
  int i = blockIdx.x * blockDim.x + threadIdx.x;
  if (i >= nd) return;
  float px = pd[3*i+0], py = pd[3*i+1], pz = pd[3*i+2];
  float bd[KNN_MAX]; int bi[KNN_MAX];
  for (int j = 0; j < k; ++j) { bd[j] = 3.0e38f; bi[j] = 0; }
  for (int s = 0; s < ns; ++s) {
    float dx = ps[3*s+0]-px, dy = ps[3*s+1]-py, dz = ps[3*s+2]-pz;
    float d2 = dx*dx + dy*dy + dz*dz;
    if (exclude_self && d2 < 1e-12f) d2 += 1e6f;
    if (d2 < bd[k-1]) {
      int j = k - 1;
      while (j > 0 && bd[j-1] > d2) { bd[j] = bd[j-1]; bi[j] = bi[j-1]; --j; }
      bd[j] = d2; bi[j] = s;
    }
  }
  for (int j = 0; j < k; ++j) {
    int e = i * k + j, s = bi[j];
    float d = sqrtf(fmaxf(bd[j], 1e-12f));
    idx[e] = s; msk[e] = (d < rmax) ? 1 : 0; dst[e] = d;
    float inv = 1.f / fmaxf(d, 1e-9f);
    unt[3*e+0] = (ps[3*s+0]-px)*inv;
    unt[3*e+1] = (ps[3*s+1]-py)*inv;
    unt[3*e+2] = (ps[3*s+2]-pz)*inv;
  }
}

// ---------------------------------------------------------------------------
// Spherical harmonics (l<=3) + gaussian/cosine-cutoff radial basis, per edge.
// ---------------------------------------------------------------------------
__global__ void edge_geom_kernel(const float* __restrict__ unt, const float* __restrict__ dst,
                                 int E, float rmax,
                                 float* __restrict__ sh, float* __restrict__ rbf) {
  int e = blockIdx.x * blockDim.x + threadIdx.x;
  if (e >= E) return;
  float x = unt[3*e], y = unt[3*e+1], z = unt[3*e+2];
  float x2 = x*x, y2 = y*y, z2 = z*z;
  float* o = sh + (size_t)e * SH_DIM;
  o[0]  = 0.28209479177387814f;
  o[1]  = 0.4886025119029199f * y;
  o[2]  = 0.4886025119029199f * z;
  o[3]  = 0.4886025119029199f * x;
  o[4]  = 1.0925484305920792f * x * y;
  o[5]  = 1.0925484305920792f * y * z;
  o[6]  = 0.31539156525252005f * (3.f*z2 - 1.f);
  o[7]  = 1.0925484305920792f * x * z;
  o[8]  = 0.5462742152960396f * (x2 - y2);
  o[9]  = 0.5900435899266435f * y * (3.f*x2 - y2);
  o[10] = 2.890611442640554f * x * y * z;
  o[11] = 0.4570457994644658f * y * (5.f*z2 - 1.f);
  o[12] = 0.3731763325901154f * z * (5.f*z2 - 3.f);
  o[13] = 0.4570457994644658f * x * (5.f*z2 - 1.f);
  o[14] = 1.445305721320277f * z * (x2 - y2);
  o[15] = 0.5900435899266435f * x * (x2 - 3.f*y2);
  float d = dst[e];
  float width = rmax / (float)(NB - 1);
  float cut = 0.5f * (cosf(PI_F * fminf(fmaxf(d / rmax, 0.f), 1.f)) + 1.f);
  for (int b = 0; b < NB; ++b) {
    float c = rmax * (float)b / (float)(NB - 1);
    float t = (d - c) / width;
    rbf[(size_t)e * NB + b] = expf(-t*t) * cut;
  }
}

// ---------------------------------------------------------------------------
// Fused SE(3)-transformer layer. One block (6 waves = 6 column tiles) per point.
// Branchless packed A-fill into pre-swizzled LDS tile; swizzled B from HBM.
// ---------------------------------------------------------------------------
__global__ __launch_bounds__(192)
void se3t_layer_kernel(const float* __restrict__ x, int Din,
                       const int* __restrict__ idx, const int* __restrict__ msk,
                       const float* __restrict__ sh, const float* __restrict__ rbf,
                       const _Float16* __restrict__ WkSw, const _Float16* __restrict__ WvSw,
                       const float* __restrict__ Wq, const float* __restrict__ Wgk,
                       const float* __restrict__ Wgv, const float* __restrict__ Wself,
                       int skip, float* __restrict__ xout) {
  __shared__ float    xc[EMB];
  __shared__ float    xn_s[16][EMB];
  __shared__ float    sh_s[16][SH_DIM];
  __shared__ float    rbf_s[16][NB];
  __shared__ __align__(32) _Float16 As[32][16];   // fragment-order A tile
  __shared__ float    qs[EMB];
  __shared__ float    ks[16][96];
  __shared__ float    vs[16][96];
  __shared__ float    lg[K_MSG];
  __shared__ float    aw[K_MSG];
  __shared__ float    outs[EMB];
  __shared__ float    red[1];

  int n = blockIdx.x, tid = threadIdx.x;
  int lane = tid & 31, wave = tid >> 5;

  const int CD  = Din * SH_DIM;
  const int nch = (CD + 31) >> 5;
  const int fpad = nch * 2;              // highest f index touched + 1 (<= EMB)

  for (int t = tid; t < Din; t += 192) xc[t] = x[(size_t)n * Din + t];
  for (int t = tid; t < 16 * Din; t += 192) {
    int r = t / Din, f = t - r * Din;
    xn_s[r][f] = (r < K_MSG) ? x[(size_t)idx[n*K_MSG + r] * Din + f] : 0.f;
  }
  if (fpad > Din) {                      // zero-pad so the A-fill is branchless
    int span = fpad - Din;
    for (int t = tid; t < 16 * span; t += 192) {
      int r = t / span, f = Din + (t - r * span);
      xn_s[r][f] = 0.f;
    }
  }
  for (int t = tid; t < 16 * SH_DIM; t += 192) {
    int r = t >> 4, s = t & 15;
    sh_s[r][s] = (r < K_MSG) ? sh[(size_t)(n*K_MSG + r) * SH_DIM + s] : 0.f;
  }
  for (int t = tid; t < 16 * NB; t += 192) {
    int r = t / NB, b = t - r * NB;
    rbf_s[r][b] = (r < K_MSG) ? rbf[(size_t)(n*K_MSG + r) * NB + b] : 0.f;
  }
  __syncthreads();

  for (int d = tid; d < EMB; d += 192) {
    float acc = 0.f;
    for (int f = 0; f < Din; ++f) acc += xc[f] * Wq[f*EMB + d];
    qs[d] = acc;
  }

  int col = wave * 16 + (lane & 15);
  const _Float16* bkBase = WkSw + ((size_t)wave * nch * 32 + lane) * 16;
  const _Float16* bvBase = WvSw + ((size_t)wave * nch * 32 + lane) * 16;
  v8f accK = {}; v8f accV = {};

  for (int ch = 0; ch < nch; ++ch) {
    __syncthreads();
    // 256 packed pairs over 192 threads: wave-uniform remainder, no mask loop
    a_fill_pair(As, xn_s, sh_s, ch, tid);
    if (tid < 64) a_fill_pair(As, xn_s, sh_s, ch, tid + 192);
    __syncthreads();
    const _Float16* bkp = bkBase + (size_t)ch * 512;
    const _Float16* bvp = bvBase + (size_t)ch * 512;
    __builtin_prefetch((const void*)(bkp + 512), 0, 3);
    __builtin_prefetch((const void*)(bvp + 512), 0, 3);
    v16h a  = *(const v16h*)(&As[lane][0]);
    v16h bk = *(const v16h*)bkp;
    v16h bv = *(const v16h*)bvp;
    accK = __builtin_amdgcn_wmma_f32_16x16x32_f16(false, a, false, bk, (short)0, accK, false, false);
    accV = __builtin_amdgcn_wmma_f32_16x16x32_f16(false, a, false, bv, (short)0, accV, false, false);
  }

  if (col < EMB) {
    #pragma unroll
    for (int r = 0; r < 8; ++r) {
      int row = r + (lane >> 4) * 8;
      float gk = 0.f, gv = 0.f;
      for (int b = 0; b < NB; ++b) {
        float rb = rbf_s[row][b];
        gk += rb * Wgk[b*EMB + col];
        gv += rb * Wgv[b*EMB + col];
      }
      ks[row][col] = gk * accK[r];
      vs[row][col] = gv * accV[r];
    }
  }
  __syncthreads();

  for (int e = wave; e < K_MSG; e += 6) {
    float p = 0.f;
    for (int d = lane; d < EMB; d += 32) p += qs[d] * ks[e][d];
    p = wave_sum(p);
    if (lane == 0) lg[e] = msk[n*K_MSG + e] ? p * INV_SQRT_EMB : -1e9f;
  }
  __syncthreads();
  if (tid == 0) {
    float mx = -3.0e38f;
    for (int e = 0; e < K_MSG; ++e) mx = fmaxf(mx, lg[e]);
    float sum = 0.f, ex[K_MSG];
    for (int e = 0; e < K_MSG; ++e) { ex[e] = expf(lg[e] - mx); sum += ex[e]; }
    for (int e = 0; e < K_MSG; ++e) aw[e] = (ex[e] / sum) * (msk[n*K_MSG + e] ? 1.f : 0.f);
  }
  __syncthreads();
  for (int d = tid; d < EMB; d += 192) {
    float o = 0.f;
    for (int e = 0; e < K_MSG; ++e) o += aw[e] * vs[e][d];
    float sf = 0.f;
    for (int f = 0; f < Din; ++f) sf += xc[f] * Wself[f*EMB + d];
    o += sf;
    if (skip) o += xc[d];
    outs[d] = o;
  }
  __syncthreads();
  if (tid < 32) {
    float ss = 0.f;
    for (int d = lane; d < EMB; d += 32) ss += outs[d] * outs[d];
    ss = wave_sum(ss);
    if (lane == 0) red[0] = rsqrtf(ss / (float)EMB + 1e-6f);
  }
  __syncthreads();
  for (int d = tid; d < EMB; d += 192) xout[(size_t)n * EMB + d] = outs[d] * red[0];
}

// ---------------------------------------------------------------------------
// Fused equivariant tensor-field evaluation at one query per block.
// 24 neighbors -> two 16-row M-tiles; WMMA accumulation; gated masked mean;
// RMS normalize; optional 1-d output linear. nTile = ceil(Dout/16).
// ---------------------------------------------------------------------------
__global__ __launch_bounds__(192)
void tensor_field_kernel(const float* __restrict__ feat,
                         const int* __restrict__ idx, const int* __restrict__ msk,
                         const float* __restrict__ sh, const float* __restrict__ rbf,
                         const _Float16* __restrict__ WSw, const float* __restrict__ Wg,
                         int Dout, int nTile,
                         const float* __restrict__ Wlin, const float* __restrict__ bias,
                         float* __restrict__ out_field, float* __restrict__ out_scalar) {
  __shared__ float    xn_s[16][EMB];
  __shared__ float    sh_s[16][SH_DIM];
  __shared__ float    rbf_s[16][NB];
  __shared__ __align__(32) _Float16 As[32][16];
  __shared__ float    fld[96];
  __shared__ float    red[2];

  int q = blockIdx.x, tid = threadIdx.x;
  int lane = tid & 31, wave = tid >> 5;
  int col = wave * 16 + (lane & 15);

  for (int t = tid; t < 96; t += 192) fld[t] = 0.f;

  const int CD  = EMB * SH_DIM;     // 1376, exact multiple of 32
  const int nch = CD >> 5;          // 43
  const _Float16* bBase = WSw + ((size_t)wave * nch * 32 + lane) * 16;

  for (int mt = 0; mt < 2; ++mt) {
    __syncthreads();
    for (int t = tid; t < 16 * EMB; t += 192) {
      int r = t / EMB, f = t - r * EMB, e = mt*16 + r;
      xn_s[r][f] = (e < K_FLD) ? feat[(size_t)idx[q*K_FLD + e] * EMB + f] : 0.f;
    }
    for (int t = tid; t < 16 * SH_DIM; t += 192) {
      int r = t >> 4, s = t & 15, e = mt*16 + r;
      sh_s[r][s] = (e < K_FLD) ? sh[(size_t)(q*K_FLD + e) * SH_DIM + s] : 0.f;
    }
    for (int t = tid; t < 16 * NB; t += 192) {
      int r = t / NB, b = t - r * NB, e = mt*16 + r;
      rbf_s[r][b] = (e < K_FLD) ? rbf[(size_t)(q*K_FLD + e) * NB + b] : 0.f;
    }
    v8f acc = {};
    for (int ch = 0; ch < nch; ++ch) {
      __syncthreads();
      a_fill_pair(As, xn_s, sh_s, ch, tid);
      if (tid < 64) a_fill_pair(As, xn_s, sh_s, ch, tid + 192);
      __syncthreads();
      if (wave < nTile) {               // uniform per-wave; barriers stay outside
        const _Float16* bp = bBase + (size_t)ch * 512;
        __builtin_prefetch((const void*)(bp + 512), 0, 3);
        v16h a = *(const v16h*)(&As[lane][0]);
        v16h b = *(const v16h*)bp;
        acc = __builtin_amdgcn_wmma_f32_16x16x32_f16(false, a, false, b, (short)0, acc, false, false);
      }
    }
    if (col < Dout) {
      #pragma unroll
      for (int r = 0; r < 8; ++r) {
        int row = r + (lane >> 4) * 8;
        int e = mt*16 + row;
        if (e < K_FLD && msk[q*K_FLD + e]) {
          float g = 0.f;
          for (int b = 0; b < NB; ++b) g += rbf_s[row][b] * Wg[b*Dout + col];
          atomicAdd(&fld[col], g * acc[r]);
        }
      }
    }
  }
  __syncthreads();
  if (tid == 0) {
    int c = 0;
    for (int e = 0; e < K_FLD; ++e) c += msk[q*K_FLD + e] ? 1 : 0;
    red[1] = fmaxf((float)c, 1.f);
  }
  __syncthreads();
  for (int d = tid; d < Dout; d += 192) fld[d] /= red[1];
  __syncthreads();
  if (tid < 32) {
    float ss = 0.f;
    for (int d = lane; d < Dout; d += 32) ss += fld[d] * fld[d];
    ss = wave_sum(ss);
    if (lane == 0) red[0] = rsqrtf(ss / (float)Dout + 1e-6f);
  }
  __syncthreads();
  if (out_field) {
    for (int d = tid; d < Dout; d += 192) out_field[(size_t)q * Dout + d] = fld[d] * red[0];
  }
  if (out_scalar && tid == 0) {
    float s = bias ? bias[0] : 0.f;
    for (int d = 0; d < Dout; ++d) s += fld[d] * red[0] * Wlin[d];
    out_scalar[q] = s;
  }
}

// ---------------------------------------------------------------------------
// Greedy clustering: 64 sequential argmax/suppress iterations, single block.
// ---------------------------------------------------------------------------
__global__ void cluster_kernel(const float* __restrict__ wlogit,
                               const int* __restrict__ rg_idx, const int* __restrict__ rg_msk,
                               const float* __restrict__ pos, float* __restrict__ qpts) {
  __shared__ float w[N_PTS];
  __shared__ float rv[256];
  __shared__ int   ri[256];
  int tid = threadIdx.x;
  for (int t = tid; t < N_PTS; t += 256) w[t] = wlogit[t];
  __syncthreads();
  for (int it = 0; it < N_QUERY; ++it) {
    float best = -3.0e38f; int bidx = N_PTS;
    for (int t = tid; t < N_PTS; t += 256) {
      float v = w[t];
      if (v > best || (v == best && t < bidx)) { best = v; bidx = t; }
    }
    rv[tid] = best; ri[tid] = bidx;
    __syncthreads();
    for (int s = 128; s > 0; s >>= 1) {
      if (tid < s) {
        if (rv[tid+s] > rv[tid] || (rv[tid+s] == rv[tid] && ri[tid+s] < ri[tid])) {
          rv[tid] = rv[tid+s]; ri[tid] = ri[tid+s];
        }
      }
      __syncthreads();
    }
    if (tid == 0) {
      int c = ri[0];
      int mem[K_RG+1]; int mv[K_RG+1]; float ls[K_RG+1];
      for (int j = 0; j < K_RG; ++j) {
        int m = rg_idx[c*K_RG + j];
        mem[j] = m;
        mv[j] = (rg_msk[c*K_RG + j] && (w[m] > -1e3f)) ? 1 : 0;
      }
      mem[K_RG] = c; mv[K_RG] = 1;
      float mx = -3.0e38f;
      for (int j = 0; j <= K_RG; ++j) { ls[j] = mv[j] ? wlogit[mem[j]] : -1e9f; mx = fmaxf(mx, ls[j]); }
      float sum = 0.f;
      for (int j = 0; j <= K_RG; ++j) { ls[j] = expf(ls[j] - mx); sum += ls[j]; }
      float qx = 0.f, qy = 0.f, qz = 0.f;
      for (int j = 0; j <= K_RG; ++j) {
        float a = ls[j] / sum;
        qx += a * pos[3*mem[j]+0]; qy += a * pos[3*mem[j]+1]; qz += a * pos[3*mem[j]+2];
      }
      qpts[3*it+0] = qx; qpts[3*it+1] = qy; qpts[3*it+2] = qz;
      for (int j = 0; j <= K_RG; ++j) if (mv[j]) w[mem[j]] = fminf(w[mem[j]], -1e4f);
    }
    __syncthreads();
  }
}

__global__ void softmax64_kernel(const float* __restrict__ logit, float* __restrict__ out) {
  __shared__ float s[N_QUERY];
  int t = threadIdx.x;
  s[t] = logit[t];
  __syncthreads();
  if (t == 0) {
    float mx = -3.0e38f;
    for (int i = 0; i < N_QUERY; ++i) mx = fmaxf(mx, s[i]);
    float sum = 0.f;
    for (int i = 0; i < N_QUERY; ++i) { s[i] = expf(s[i] - mx); sum += s[i]; }
    for (int i = 0; i < N_QUERY; ++i) s[i] /= sum;
  }
  __syncthreads();
  out[t] = s[t];
}

// ---------------------------------------------------------------------------
extern "C" void kernel_launch(void* const* d_in, const int* in_sizes, int n_in,
                              void* d_out, int out_size, void* d_ws, size_t ws_size,
                              hipStream_t stream) {
  (void)in_sizes; (void)n_in; (void)out_size; (void)ws_size;

  const float* pos     = (const float*)d_in[0];
  const float* feature = (const float*)d_in[1];
  int ii = 2;
  const float *Wq[7], *Wkf[7], *Wvf[7], *Wgk[7], *Wgv[7], *Wself[7];
  for (int l = 0; l < 7; ++l) {
    Wq[l]    = (const float*)d_in[ii++];
    Wkf[l]   = (const float*)d_in[ii++];
    Wvf[l]   = (const float*)d_in[ii++];
    Wgk[l]   = (const float*)d_in[ii++];
    Wgv[l]   = (const float*)d_in[ii++];
    Wself[l] = (const float*)d_in[ii++];
  }
  const float* wfW    = (const float*)d_in[ii++];
  const float* wfWg   = (const float*)d_in[ii++];
  const float* wfWlin = (const float*)d_in[ii++];
  const float* wfb    = (const float*)d_in[ii++];
  const float* ffW    = (const float*)d_in[ii++];
  const float* ffWg   = (const float*)d_in[ii++];

  char* ws = (char*)d_ws;
  size_t off = 0;
  auto alloc = [&](size_t bytes) -> void* {
    off = (off + 255) & ~(size_t)255;
    void* p = ws + off;
    off += bytes;
    return p;
  };

  const int EM = N_PTS * K_MSG;     // 49152
  const int EF = N_PTS * K_FLD;     // 98304
  const int ER = N_PTS * K_RG;      // 65536
  const int EQ = N_QUERY * K_FLD;   // 1536
  const int NCH_FULL = (EMB * SH_DIM) >> 5;   // 43

  // swizzled f16 weights: [nTile][nch][32 lanes][16] halfs
  _Float16 *WkSw[7], *WvSw[7];
  int nchL[7];
  for (int l = 0; l < 7; ++l) {
    int din = l ? EMB : 3;
    int cd = din * SH_DIM;
    nchL[l] = (cd + 31) >> 5;
    size_t n = (size_t)6 * nchL[l] * 512;
    WkSw[l] = (_Float16*)alloc(n * sizeof(_Float16));
    WvSw[l] = (_Float16*)alloc(n * sizeof(_Float16));
  }
  _Float16* wfWSw = (_Float16*)alloc((size_t)1 * NCH_FULL * 512 * sizeof(_Float16));
  _Float16* ffWSw = (_Float16*)alloc((size_t)6 * NCH_FULL * 512 * sizeof(_Float16));

  int   *idxM = (int*)alloc(EM*4),   *mskM = (int*)alloc(EM*4);
  float *dstM = (float*)alloc(EM*4), *untM = (float*)alloc((size_t)EM*3*4);
  float *shM  = (float*)alloc((size_t)EM*SH_DIM*4), *rbM = (float*)alloc((size_t)EM*NB*4);

  int   *idxF = (int*)alloc(EF*4),   *mskF = (int*)alloc(EF*4);
  float *dstF = (float*)alloc(EF*4), *untF = (float*)alloc((size_t)EF*3*4);
  float *shF  = (float*)alloc((size_t)EF*SH_DIM*4), *rbF = (float*)alloc((size_t)EF*NB*4);

  int   *idxR = (int*)alloc(ER*4),   *mskR = (int*)alloc(ER*4);
  float *dstR = (float*)alloc(ER*4), *untR = (float*)alloc((size_t)ER*3*4);

  int   *idxQ = (int*)alloc(EQ*4),   *mskQ = (int*)alloc(EQ*4);
  float *dstQ = (float*)alloc(EQ*4), *untQ = (float*)alloc((size_t)EQ*3*4);
  float *shQ  = (float*)alloc((size_t)EQ*SH_DIM*4), *rbQ = (float*)alloc((size_t)EQ*NB*4);

  float *xb0  = (float*)alloc((size_t)N_PTS*EMB*4);
  float *xb1  = (float*)alloc((size_t)N_PTS*EMB*4);
  float *wlog = (float*)alloc(N_PTS*4);
  float *qlog = (float*)alloc(N_QUERY*4);

  // ---- swizzle weights into WMMA B-fragment order ----
  for (int l = 0; l < 7; ++l) {
    int din = l ? EMB : 3;
    int cd = din * SH_DIM;
    int total = 6 * nchL[l] * 512;
    swizzle_b_kernel<<<(total + 255) / 256, 256, 0, stream>>>(Wkf[l], WkSw[l], cd, EMB, 6, nchL[l]);
    swizzle_b_kernel<<<(total + 255) / 256, 256, 0, stream>>>(Wvf[l], WvSw[l], cd, EMB, 6, nchL[l]);
  }
  {
    int total = 1 * NCH_FULL * 512;
    swizzle_b_kernel<<<(total + 255) / 256, 256, 0, stream>>>(wfW, wfWSw, EMB*SH_DIM, 10, 1, NCH_FULL);
    total = 6 * NCH_FULL * 512;
    swizzle_b_kernel<<<(total + 255) / 256, 256, 0, stream>>>(ffW, ffWSw, EMB*SH_DIM, EMB, 6, NCH_FULL);
  }

  // ---- message-passing graph + geometry ----
  knn_kernel<<<(N_PTS + 63) / 64, 64, 0, stream>>>(pos, pos, N_PTS, N_PTS, K_MSG,
                                                   MAX_RADIUS_F, 1, idxM, mskM, dstM, untM);
  edge_geom_kernel<<<(EM + 127) / 128, 128, 0, stream>>>(untM, dstM, EM, MAX_RADIUS_F, shM, rbM);

  // ---- 7 SE(3)-transformer layers ----
  const float* xin = feature;
  float* xout = xb0;
  for (int l = 0; l < 7; ++l) {
    se3t_layer_kernel<<<N_PTS, 192, 0, stream>>>(xin, l ? EMB : 3, idxM, mskM, shM, rbM,
                                                 WkSw[l], WvSw[l], Wq[l], Wgk[l], Wgv[l],
                                                 Wself[l], l ? 1 : 0, xout);
    xin = xout;
    xout = (xout == xb0) ? xb1 : xb0;
  }
  const float* feat = xin;

  // ---- weight field at all points -> wlogit ----
  knn_kernel<<<(N_PTS + 63) / 64, 64, 0, stream>>>(pos, pos, N_PTS, N_PTS, K_FLD,
                                                   FIELD_CUTOFF_F, 1, idxF, mskF, dstF, untF);
  edge_geom_kernel<<<(EF + 127) / 128, 128, 0, stream>>>(untF, dstF, EF, FIELD_CUTOFF_F, shF, rbF);
  tensor_field_kernel<<<N_PTS, 192, 0, stream>>>(feat, idxF, mskF, shF, rbF, wfWSw, wfWg, 10, 1,
                                                 wfWlin, wfb, nullptr, wlog);

  // ---- clustering graph + greedy clustering -> query points ----
  knn_kernel<<<(N_PTS + 63) / 64, 64, 0, stream>>>(pos, pos, N_PTS, N_PTS, K_RG,
                                                   QUERY_RADIUS_F * 0.999f, 1, idxR, mskR, dstR, untR);
  float* fo   = (float*)d_out;
  float* qpts = fo + N_QUERY * EMB;                 // query_points output slot
  float* qatt = fo + N_QUERY * EMB + N_QUERY * 3;   // query_attention output slot
  cluster_kernel<<<1, 256, 0, stream>>>(wlog, idxR, mskR, pos, qpts);

  // ---- feature / weight fields at query points ----
  knn_kernel<<<1, 64, 0, stream>>>(qpts, pos, N_QUERY, N_PTS, K_FLD,
                                   FIELD_CUTOFF_F, 0, idxQ, mskQ, dstQ, untQ);
  edge_geom_kernel<<<(EQ + 127) / 128, 128, 0, stream>>>(untQ, dstQ, EQ, FIELD_CUTOFF_F, shQ, rbQ);
  tensor_field_kernel<<<N_QUERY, 192, 0, stream>>>(feat, idxQ, mskQ, shQ, rbQ, ffWSw, ffWg, EMB, 6,
                                                   nullptr, nullptr, fo, nullptr);
  tensor_field_kernel<<<N_QUERY, 192, 0, stream>>>(feat, idxQ, mskQ, shQ, rbQ, wfWSw, wfWg, 10, 1,
                                                   wfWlin, wfb, nullptr, qlog);
  softmax64_kernel<<<1, N_QUERY, 0, stream>>>(qlog, qatt);
}
